// TCC_42528766165259
// MI455X (gfx1250) — matrix operand order repaired and checked by reading, not compiled, not run on hardware
//
#include <hip/hip_runtime.h>
#include <hip/hip_bf16.h>
#include <stdint.h>

typedef __attribute__((ext_vector_type(16))) __bf16 v16bf;
typedef __attribute__((ext_vector_type(8)))  __bf16 v8bf;
typedef __attribute__((ext_vector_type(8)))  float  v8f;

#define DEVINL __device__ __forceinline__

constexpr int   kN      = 4096;
constexpr int   kC      = 256;
constexpr int   kJP     = 8;            // j-partitions for streaming kernels
constexpr int   kJTile  = kN / kJP;     // 512 columns per partition
constexpr int   kCS     = 4;            // c-slices in k_softnn
constexpr float kScale  = 1.0f / (256.0f * 0.1f);   // 1/(C*TEMPERATURE)
constexpr float kLambda = 0.001f;

DEVINL v8f wmma_bf16(v16bf a, v16bf b, v8f c) {
  // D = A(16x32) * B(32x16) + C, f32 accumulate
  return __builtin_amdgcn_wmma_f32_16x16x32_bf16(false, a, false, b, (short)0, c,
                                                 false, false);
}

// Async DMA: copy 32 contiguous bytes global -> LDS (2 x b128, ASYNCcnt-tracked).
// INST_OFFSET is applied to both the LDS and the global address (ISA 15.18).
DEVINL void async_stage32(uint32_t lds_off, const __bf16* gptr) {
  asm volatile(
      "global_load_async_to_lds_b128 %0, %1, off\n\t"
      "global_load_async_to_lds_b128 %0, %1, off offset:16"
      :: "v"(lds_off), "v"(gptr) : "memory");
}

DEVINL void wait_async_zero() {
  asm volatile("s_wait_asynccnt 0x0" ::: "memory");
}

// A-operand tile: rows row0..row0+15, K = k0..k0+31, row-major bf16 [ldk].
// ISA 16-bit A layout: lane m holds K {koff..koff+7} and {16+koff..+7}, koff=8*(lane>=16).
DEVINL v16bf load_a_tile(const __bf16* __restrict__ base, int row0, int k0,
                         int ldk, int lane) {
  const int r    = row0 + (lane & 15);
  const int koff = (lane >> 4) << 3;
  const __bf16* p = base + (size_t)r * ldk + k0 + koff;
  v8bf c0 = *(const v8bf*)p;
  v8bf c1 = *(const v8bf*)(p + 16);
  v16bf v;
#pragma unroll
  for (int t = 0; t < 8; ++t) { v[t] = c0[t]; v[t + 8] = c1[t]; }
  return v;
}

// B-operand tile: B[k][n] = base[(n0+n)*ldk + k0 + k], K=32, N=16.
// ISA 16-bit B layout: lane n holds K {koff..koff+7} and {16+koff..+7}, koff=8*(lane>=16).
DEVINL v16bf load_b_tile(const __bf16* __restrict__ base, int n0, int k0,
                         int ldk, int lane) {
  const int r  = n0 + (lane & 15);
  const int kb = (lane >> 4) << 3;
  const __bf16* p = base + (size_t)r * ldk + k0 + kb;
  v8bf c0 = *(const v8bf*)p;
  v8bf c1 = *(const v8bf*)(p + 16);
  v16bf v;
#pragma unroll
  for (int t = 0; t < 8; ++t) { v[t] = c0[t]; v[t + 8] = c1[t]; }
  return v;
}

// B-operand tile from an LDS-staged [16 x 256] row-major chunk.
DEVINL v16bf lds_b_tile(const __bf16* base, int k0, int lane) {
  const int r  = lane & 15;
  const int kb = (lane >> 4) << 3;
  const __bf16* p = base + r * kC + k0 + kb;
  v8bf c0 = *(const v8bf*)p;
  v8bf c1 = *(const v8bf*)(p + 16);
  v16bf v;
#pragma unroll
  for (int t = 0; t < 8; ++t) { v[t] = c0[t]; v[t + 8] = c1[t]; }
  return v;
}

// Sum across the 16 lanes of each wave32 half (xor masks stay inside halves).
DEVINL float hreduce16(float x) {
  x += __shfl_xor(x, 1, 32);
  x += __shfl_xor(x, 2, 32);
  x += __shfl_xor(x, 4, 32);
  x += __shfl_xor(x, 8, 32);
  return x;
}

__global__ void k_zero(float* out) {
  if (threadIdx.x == 0 && blockIdx.x == 0) out[0] = 0.0f;
}

// Split fp32 -> bf16 hi/lo, row norms, and transposed-hi copy (for the P@V matmul).
__global__ __launch_bounds__(256) void k_prep(
    const float* __restrict__ a, const float* __restrict__ b,
    __bf16* __restrict__ a_hi, __bf16* __restrict__ a_lo,
    __bf16* __restrict__ b_hi, __bf16* __restrict__ b_lo,
    __bf16* __restrict__ aT_hi, __bf16* __restrict__ bT_hi,
    float* __restrict__ na, float* __restrict__ nb) {
  const int blk = blockIdx.x;
  const float* src; __bf16 *hi, *lo, *T; float* nrm; int row0;
  if (blk < kN / 16) { src = a; hi = a_hi; lo = a_lo; T = aT_hi; nrm = na; row0 = blk * 16; }
  else { src = b; hi = b_hi; lo = b_lo; T = bT_hi; nrm = nb; row0 = (blk - kN / 16) * 16; }
  const int t  = threadIdx.x;
  const int r  = row0 + (t >> 4);
  const int c0 = (t & 15) * 16;
  const float* p = src + (size_t)r * kC + c0;
  float nsum = 0.f;
  v8bf h8[2], l8[2];
#pragma unroll
  for (int g = 0; g < 2; ++g) {
#pragma unroll
    for (int i = 0; i < 8; ++i) {
      float x = p[g * 8 + i];
      __bf16 h = (__bf16)x;
      h8[g][i] = h;
      l8[g][i] = (__bf16)(x - (float)h);
      T[(size_t)(c0 + g * 8 + i) * kN + r] = h;
      nsum += x * x;
    }
  }
  *(v8bf*)(hi + (size_t)r * kC + c0)     = h8[0];
  *(v8bf*)(hi + (size_t)r * kC + c0 + 8) = h8[1];
  *(v8bf*)(lo + (size_t)r * kC + c0)     = l8[0];
  *(v8bf*)(lo + (size_t)r * kC + c0 + 8) = l8[1];
  nsum = hreduce16(nsum);
  if ((t & 15) == 0) nrm[r] = nsum;
}

// e_ij = exp((2 x_i.y_j - |x_i|^2 - |y_j|^2)*scale)  (logits <= 0 => no max pass).
// Writes P (bf16) and per-partition row sums. y-chunks are double-buffered in LDS
// via async-to-LDS DMA, shared by all 8 waves of the block. Two independent WMMA
// accumulator chains (even/odd k-step) keep the XDL fed across LDS waits.
__global__ __launch_bounds__(256) void k_simexp(
    const __bf16* __restrict__ xhi, const __bf16* __restrict__ xlo,
    const float* __restrict__ nx,
    const __bf16* __restrict__ yhi, const __bf16* __restrict__ ylo,
    const float* __restrict__ ny,
    __bf16* __restrict__ Pout, float* __restrict__ l_part) {
  __shared__ __align__(16) __bf16 ybuf[2][2][16 * kC];   // [buf][hi/lo]
  const int lane  = threadIdx.x & 31;
  const int wave  = threadIdx.x >> 5;
  const int rb    = blockIdx.x / kJP;
  const int part  = blockIdx.x % kJP;
  const int row0  = rb * 128 + wave * 16;
  const int jbase = part * kJTile;
  const int h8    = (lane >> 4) * 8;

  v16bf axh[8], axl[8];
#pragma unroll
  for (int ks = 0; ks < 8; ++ks) {
    axh[ks] = load_a_tile(xhi, row0, ks * 32, kC, lane);
    axl[ks] = load_a_tile(xlo, row0, ks * 32, kC, lane);
  }
  float nx8[8];
#pragma unroll
  for (int v = 0; v < 8; ++v) nx8[v] = nx[row0 + h8 + v];

  const int tr = threadIdx.x >> 4;          // staging: row 0..15
  const int tc = (threadIdx.x & 15) * 16;   // staging: 16-col chunk
  const uint32_t lds_base = (uint32_t)(uintptr_t)(&ybuf[0][0][0]);
  const uint32_t elem_off = (uint32_t)(tr * kC + tc) * 2u;
  auto stage = [&](int buf, int j0) {
    const __bf16* ph = yhi + (size_t)(j0 + tr) * kC + tc;
    const __bf16* pl = ylo + (size_t)(j0 + tr) * kC + tc;
    const uint32_t b = lds_base + (uint32_t)buf * (16u * kC * 4u) + elem_off;
    async_stage32(b, ph);
    async_stage32(b + 16u * kC * 2u, pl);
  };

  float ls[8] = {0, 0, 0, 0, 0, 0, 0, 0};
  const v8f vz = {0, 0, 0, 0, 0, 0, 0, 0};
  stage(0, jbase);
  wait_async_zero();
  __syncthreads();
#pragma unroll 1
  for (int jc = 0; jc < kJTile / 16; ++jc) {
    const int j0  = jbase + jc * 16;
    const int cur = jc & 1;
    if (jc + 1 < kJTile / 16) stage(cur ^ 1, j0 + 16);  // DMA next chunk
    v8f acc0 = vz, acc1 = vz;                           // two independent chains
#pragma unroll
    for (int ks = 0; ks < 8; ks += 2) {
      v16bf byh0 = lds_b_tile(&ybuf[cur][0][0], ks * 32, lane);
      v16bf byl0 = lds_b_tile(&ybuf[cur][1][0], ks * 32, lane);
      v16bf byh1 = lds_b_tile(&ybuf[cur][0][0], (ks + 1) * 32, lane);
      v16bf byl1 = lds_b_tile(&ybuf[cur][1][0], (ks + 1) * 32, lane);
      acc0 = wmma_bf16(axh[ks], byh0, acc0);
      acc1 = wmma_bf16(axh[ks + 1], byh1, acc1);
      acc0 = wmma_bf16(axl[ks], byh0, acc0);
      acc1 = wmma_bf16(axl[ks + 1], byh1, acc1);
      acc0 = wmma_bf16(axh[ks], byl0, acc0);
      acc1 = wmma_bf16(axh[ks + 1], byl1, acc1);
    }
    v8f acc = acc0 + acc1;
    const float nyn = ny[j0 + (lane & 15)];
#pragma unroll
    for (int v = 0; v < 8; ++v) {
      float e = __expf((2.f * acc[v] - nx8[v] - nyn) * kScale);
      ls[v] += e;
      Pout[(size_t)(row0 + h8 + v) * kN + j0 + (lane & 15)] = (__bf16)e;
    }
    wait_async_zero();   // our DMA into buf^1 has landed
    __syncthreads();     // everyone done reading cur / writing cur^1
  }
#pragma unroll
  for (int v = 0; v < 8; ++v) ls[v] = hreduce16(ls[v]);
  if ((lane & 15) == 0) {
#pragma unroll
    for (int v = 0; v < 8; ++v) l_part[part * kN + row0 + h8 + v] = ls[v];
  }
}

// Fixed-order reduction of the partition partial sums (deterministic).
__global__ __launch_bounds__(256) void k_reduce_l(const float* __restrict__ l_part,
                                                  float* __restrict__ l) {
  const int r = blockIdx.x * 256 + threadIdx.x;
  float s = 0.f;
#pragma unroll
  for (int p = 0; p < kJP; ++p) s += l_part[p * kN + r];
  l[r] = s;
}

// soft_nn = (P @ Y) / l : pure WMMA matmul over the materialized P,
// 4 independent accumulator chains, software-pipelined against L2 latency.
__global__ __launch_bounds__(256) void k_softnn(
    const __bf16* __restrict__ P, const __bf16* __restrict__ yT_hi,
    const float* __restrict__ lsum, float* __restrict__ sOut) {
  const int lane = threadIdx.x & 31;
  const int wave = threadIdx.x >> 5;
  const int rb   = blockIdx.x / kCS;
  const int cs   = blockIdx.x % kCS;
  const int row0 = rb * 128 + wave * 16;
  const int c0   = cs * (kC / kCS);          // 64-column slice
  const int h8   = (lane >> 4) * 8;

  v8f acc[4];
  const v8f vz = {0, 0, 0, 0, 0, 0, 0, 0};
#pragma unroll
  for (int ct = 0; ct < 4; ++ct) acc[ct] = vz;

  v16bf pa = load_a_tile(P, row0, 0, kN, lane);
  v16bf bt[4];
#pragma unroll
  for (int ct = 0; ct < 4; ++ct) bt[ct] = load_b_tile(yT_hi, c0 + ct * 16, 0, kN, lane);

#pragma unroll 1
  for (int j0 = 0; j0 < kN; j0 += 32) {
    v16bf pan = pa;
    v16bf btn[4] = {bt[0], bt[1], bt[2], bt[3]};
    if (j0 + 32 < kN) {
      pan = load_a_tile(P, row0, j0 + 32, kN, lane);
#pragma unroll
      for (int ct = 0; ct < 4; ++ct)
        btn[ct] = load_b_tile(yT_hi, c0 + ct * 16, j0 + 32, kN, lane);
    }
#pragma unroll
    for (int ct = 0; ct < 4; ++ct) acc[ct] = wmma_bf16(pa, bt[ct], acc[ct]);
    pa = pan;
#pragma unroll
    for (int ct = 0; ct < 4; ++ct) bt[ct] = btn[ct];
  }
  float li8[8];
#pragma unroll
  for (int v = 0; v < 8; ++v) li8[v] = 1.0f / lsum[row0 + h8 + v];
#pragma unroll
  for (int ct = 0; ct < 4; ++ct) {
#pragma unroll
    for (int v = 0; v < 8; ++v) {
      sOut[(size_t)(row0 + h8 + v) * kC + c0 + ct * 16 + (lane & 15)] =
          acc[ct][v] * li8[v];
    }
  }
}

// Streams sim_ba over a j-partition; accumulates E / E*w / E*w^2 partials.
__global__ __launch_bounds__(256) void k_loss(
    const float* __restrict__ sIn,
    const __bf16* __restrict__ yhi, const __bf16* __restrict__ ylo,
    const float* __restrict__ ny,
    const float* __restrict__ idxv,
    float* __restrict__ e_part, float* __restrict__ s1_part,
    float* __restrict__ s2_part) {
  __shared__ __align__(16) __bf16 ybuf[2][2][16 * kC];
  const int lane  = threadIdx.x & 31;
  const int wave  = threadIdx.x >> 5;
  const int rb    = blockIdx.x / kJP;
  const int part  = blockIdx.x % kJP;
  const int row0  = rb * 128 + wave * 16;
  const int jbase = part * kJTile;
  const int h8    = (lane >> 4) * 8;

  // Preload soft_nn rows: fp32 -> split bf16 hi/lo in A-operand layout,
  // and derive the row norms from the same data (lane + lane^16 cover a row).
  v16bf axh[8], axl[8];
  float nsq = 0.f;
  {
    const int r    = row0 + (lane & 15);
    const int koff = (lane >> 4) << 3;
#pragma unroll
    for (int ks = 0; ks < 8; ++ks) {
      const float* p0 = sIn + (size_t)r * kC + ks * 32 + koff;
#pragma unroll
      for (int t = 0; t < 8; ++t) {
        float x0 = p0[t];
        __bf16 hh0 = (__bf16)x0;
        axh[ks][t] = hh0;
        axl[ks][t] = (__bf16)(x0 - (float)hh0);
        float x1 = p0[16 + t];
        __bf16 hh1 = (__bf16)x1;
        axh[ks][t + 8] = hh1;
        axl[ks][t + 8] = (__bf16)(x1 - (float)hh1);
        nsq += x0 * x0 + x1 * x1;
      }
    }
  }
  nsq += __shfl_xor(nsq, 16, 32);            // full row norm at lane (row&15)
  float nx8[8];
#pragma unroll
  for (int v = 0; v < 8; ++v) nx8[v] = __shfl(nsq, h8 + v, 32);

  const int tr = threadIdx.x >> 4;
  const int tc = (threadIdx.x & 15) * 16;
  const uint32_t lds_base = (uint32_t)(uintptr_t)(&ybuf[0][0][0]);
  const uint32_t elem_off = (uint32_t)(tr * kC + tc) * 2u;
  auto stage = [&](int buf, int j0) {
    const __bf16* ph = yhi + (size_t)(j0 + tr) * kC + tc;
    const __bf16* pl = ylo + (size_t)(j0 + tr) * kC + tc;
    const uint32_t b = lds_base + (uint32_t)buf * (16u * kC * 4u) + elem_off;
    async_stage32(b, ph);
    async_stage32(b + 16u * kC * 2u, pl);
  };

  float E[8]  = {0, 0, 0, 0, 0, 0, 0, 0};
  float S1[8] = {0, 0, 0, 0, 0, 0, 0, 0};
  float S2[8] = {0, 0, 0, 0, 0, 0, 0, 0};
  const v8f vz = {0, 0, 0, 0, 0, 0, 0, 0};
  stage(0, jbase);
  wait_async_zero();
  __syncthreads();
#pragma unroll 1
  for (int jc = 0; jc < kJTile / 16; ++jc) {
    const int j0  = jbase + jc * 16;
    const int cur = jc & 1;
    if (jc + 1 < kJTile / 16) stage(cur ^ 1, j0 + 16);
    v8f acc0 = vz, acc1 = vz;
#pragma unroll
    for (int ks = 0; ks < 8; ks += 2) {
      v16bf byh0 = lds_b_tile(&ybuf[cur][0][0], ks * 32, lane);
      v16bf byl0 = lds_b_tile(&ybuf[cur][1][0], ks * 32, lane);
      v16bf byh1 = lds_b_tile(&ybuf[cur][0][0], (ks + 1) * 32, lane);
      v16bf byl1 = lds_b_tile(&ybuf[cur][1][0], (ks + 1) * 32, lane);
      acc0 = wmma_bf16(axh[ks], byh0, acc0);
      acc1 = wmma_bf16(axh[ks + 1], byh1, acc1);
      acc0 = wmma_bf16(axl[ks], byh0, acc0);
      acc1 = wmma_bf16(axl[ks + 1], byh1, acc1);
      acc0 = wmma_bf16(axh[ks], byl0, acc0);
      acc1 = wmma_bf16(axh[ks + 1], byl1, acc1);
    }
    v8f acc = acc0 + acc1;
    const float nyn = ny[j0 + (lane & 15)];
    const float w   = idxv[j0 + (lane & 15)];
#pragma unroll
    for (int v = 0; v < 8; ++v) {
      float e = __expf((2.f * acc[v] - nx8[v] - nyn) * kScale);
      E[v]  += e;
      S1[v] += e * w;
      S2[v] += e * w * w;
    }
    wait_async_zero();
    __syncthreads();
  }
#pragma unroll
  for (int v = 0; v < 8; ++v) {
    E[v]  = hreduce16(E[v]);
    S1[v] = hreduce16(S1[v]);
    S2[v] = hreduce16(S2[v]);
  }
  if ((lane & 15) == 0) {
    const int base = part * kN + row0 + h8;
#pragma unroll
    for (int v = 0; v < 8; ++v) {
      e_part[base + v]  = E[v];
      s1_part[base + v] = S1[v];
      s2_part[base + v] = S2[v];
    }
  }
}

// Deterministic final reduction: combine partition partials, per-row loss,
// fixed-order tree sum, accumulate into the scalar output.
__global__ __launch_bounds__(256) void k_finish(
    const float* __restrict__ e_part, const float* __restrict__ s1_part,
    const float* __restrict__ s2_part, const float* __restrict__ idxv,
    float* __restrict__ out) {
  __shared__ float red[256];
  float acc = 0.f;
  for (int r = threadIdx.x; r < kN; r += 256) {
    float E = 0.f, S1 = 0.f, S2 = 0.f;
#pragma unroll
    for (int p = 0; p < kJP; ++p) {
      E  += e_part[p * kN + r];
      S1 += s1_part[p * kN + r];
      S2 += s2_part[p * kN + r];
    }
    float inv   = 1.0f / E;
    float preds = S1 * inv;
    float pv    = fmaxf(S2 * inv - preds * preds, 1e-20f);
    float d     = idxv[r] - preds;
    acc += d * d / pv + kLambda * __logf(pv);
  }
  red[threadIdx.x] = acc;
  __syncthreads();
  for (int s = 128; s > 0; s >>= 1) {
    if (threadIdx.x < s) red[threadIdx.x] += red[threadIdx.x + s];
    __syncthreads();
  }
  if (threadIdx.x == 0) out[0] += red[0] * (1.0f / (2.0f * kN));
}

extern "C" void kernel_launch(void* const* d_in, const int* in_sizes, int n_in,
                              void* d_out, int out_size, void* d_ws, size_t ws_size,
                              hipStream_t stream) {
  (void)in_sizes; (void)n_in; (void)out_size; (void)ws_size;
  const float* emb_a = (const float*)d_in[0];
  const float* emb_b = (const float*)d_in[1];
  const float* idx_a = (const float*)d_in[2];
  const float* idx_b = (const float*)d_in[3];
  float* out = (float*)d_out;

  char* w = (char*)d_ws;
  auto take = [&](size_t bytes) {
    char* p = w;
    w += (bytes + 255) & ~(size_t)255;
    return p;
  };
  const size_t mat = (size_t)kN * kC * sizeof(__bf16);
  __bf16* a_hi = (__bf16*)take(mat);
  __bf16* a_lo = (__bf16*)take(mat);
  __bf16* b_hi = (__bf16*)take(mat);
  __bf16* b_lo = (__bf16*)take(mat);
  __bf16* aT   = (__bf16*)take(mat);
  __bf16* bT   = (__bf16*)take(mat);
  float* na    = (float*)take(kN * sizeof(float));
  float* nb    = (float*)take(kN * sizeof(float));
  __bf16* P    = (__bf16*)take((size_t)kN * kN * sizeof(__bf16));  // 32 MB, L2-resident
  float* l_prt = (float*)take((size_t)kJP * kN * sizeof(float));
  float* lsum  = (float*)take(kN * sizeof(float));
  float* s     = (float*)take((size_t)kN * kC * sizeof(float));
  float* e_prt = (float*)take((size_t)kJP * kN * sizeof(float));
  float* s1prt = (float*)take((size_t)kJP * kN * sizeof(float));
  float* s2prt = (float*)take((size_t)kJP * kN * sizeof(float));

  const int RB = kN / 128;   // 32 row-blocks

  k_zero<<<1, 32, 0, stream>>>(out);
  k_prep<<<2 * kN / 16, 256, 0, stream>>>(emb_a, emb_b, a_hi, a_lo, b_hi, b_lo,
                                          aT, bT, na, nb);
  // direction A -> B
  k_simexp<<<RB * kJP, 256, 0, stream>>>(a_hi, a_lo, na, b_hi, b_lo, nb, P, l_prt);
  k_reduce_l<<<kN / 256, 256, 0, stream>>>(l_prt, lsum);
  k_softnn<<<RB * kCS, 256, 0, stream>>>(P, bT, lsum, s);
  k_loss<<<RB * kJP, 256, 0, stream>>>(s, a_hi, a_lo, na, idx_a, e_prt, s1prt, s2prt);
  k_finish<<<1, 256, 0, stream>>>(e_prt, s1prt, s2prt, idx_a, out);
  // direction B -> A
  k_simexp<<<RB * kJP, 256, 0, stream>>>(b_hi, b_lo, nb, a_hi, a_lo, na, P, l_prt);
  k_reduce_l<<<kN / 256, 256, 0, stream>>>(l_prt, lsum);
  k_softnn<<<RB * kCS, 256, 0, stream>>>(P, aT, lsum, s);
  k_loss<<<RB * kJP, 256, 0, stream>>>(s, b_hi, b_lo, nb, idx_b, e_prt, s1prt, s2prt);
  k_finish<<<1, 256, 0, stream>>>(e_prt, s1prt, s2prt, idx_b, out);
}
